// Head_81604378624165
// MI455X (gfx1250) — matrix operand order repaired and checked by reading, not compiled, not run on hardware
//
#include <hip/hip_runtime.h>

#define T_SEQ 256
#define C_EMB 384
#define H_DIM 64

typedef __attribute__((ext_vector_type(16))) __bf16 v16bf;
typedef __attribute__((ext_vector_type(8)))  __bf16 v8bf;
typedef __attribute__((ext_vector_type(8)))  float  v8f;
typedef __attribute__((ext_vector_type(4)))  float  v4f;

__device__ __forceinline__ v8f v8f_zero() {
  v8f z;
#pragma unroll
  for (int i = 0; i < 8; ++i) z[i] = 0.0f;
  return z;
}

__device__ __forceinline__ v16bf pack16(v8bf lo, v8bf hi) {
  v16bf r;
#pragma unroll
  for (int i = 0; i < 8; ++i) { r[i] = lo[i]; r[8 + i] = hi[i]; }
  return r;
}

// One block per batch element. 256 threads = 8 wave32.
__global__ __launch_bounds__(256) void attn_head_kernel(
    const float* __restrict__ x,   // [B, T, C]
    const float* __restrict__ Wq,  // [C, H]
    const float* __restrict__ Wk,  // [C, H]
    const float* __restrict__ Wv,  // [C, H]
    float* __restrict__ out)       // [B, T, H]
{
  __shared__ __align__(16) __bf16 sQ[T_SEQ * H_DIM];   // [t][h]
  __shared__ __align__(16) __bf16 sK[T_SEQ * H_DIM];   // [t][h]
  __shared__ __align__(16) __bf16 sVt[H_DIM * T_SEQ];  // [h][t] (transposed)
  union UWP {
    __bf16 w[H_DIM * C_EMB];      // W^T bf16, [h][c]  (projection phase)
    __bf16 p[8][16 * T_SEQ];      // per-wave P scratch (attention phase)
  };
  __shared__ __align__(16) UWP sU;

  const int b     = blockIdx.x;
  const int tid   = threadIdx.x;
  const int wave  = tid >> 5;
  const int lane  = tid & 31;
  const int lhalf = lane >> 4;   // 0: lanes 0-15, 1: lanes 16-31
  const int l16   = lane & 15;
  const int kloA  = lhalf * 8;   // A-fragment K start (16-bit A layout)
  const int kloB  = lhalf * 8;   // B-fragment K start (16-bit B layout)

  // ---------------- Phase 1: Q/K/V projections ----------------
#pragma unroll 1
  for (int p = 0; p < 3; ++p) {
    const float* G = (p == 0) ? Wq : (p == 1) ? Wk : Wv;
    // Stage W^T (bf16) into LDS: sU.w[h*C + c] = W[c*H + h]
    for (int i = tid; i < C_EMB * H_DIM; i += 256) {
      const int c = i >> 6;
      const int h = i & 63;
      sU.w[h * C_EMB + c] = (__bf16)G[i];
    }
    __syncthreads();

    // Uniform (per-p) store-target setup: no per-element branches below.
    __bf16* qkDst = (p == 0) ? sQ : sK;

#pragma unroll 1
    for (int r2 = 0; r2 < 2; ++r2) {
      const int mt = wave * 2 + r2;  // M-tile (16 rows of t)
      v8f acc0 = v8f_zero(), acc1 = v8f_zero(), acc2 = v8f_zero(), acc3 = v8f_zero();
      const float* xrow = x + ((size_t)b * T_SEQ + mt * 16 + l16) * C_EMB;
#pragma unroll 1
      for (int kc = 0; kc < 12; ++kc) {  // K = 384 in chunks of 32
        const float* xk = xrow + kc * 32 + kloA;
        v4f f0 = *(const v4f*)(xk);
        v4f f1 = *(const v4f*)(xk + 4);
        v4f f2 = *(const v4f*)(xk + 16);
        v4f f3 = *(const v4f*)(xk + 20);
        v16bf a;
#pragma unroll
        for (int i = 0; i < 4; ++i) {
          a[i]      = (__bf16)f0[i];
          a[4 + i]  = (__bf16)f1[i];
          a[8 + i]  = (__bf16)f2[i];
          a[12 + i] = (__bf16)f3[i];
        }
#pragma unroll
        for (int nt = 0; nt < 4; ++nt) {
          const __bf16* wrow = &sU.w[(nt * 16 + l16) * C_EMB + kc * 32 + kloB];
          v16bf bb = pack16(*(const v8bf*)wrow, *(const v8bf*)(wrow + 16));
          v8f& acc = (nt == 0) ? acc0 : (nt == 1) ? acc1 : (nt == 2) ? acc2 : acc3;
          acc = __builtin_amdgcn_wmma_f32_16x16x32_bf16(false, a, false, bb,
                                                        (short)0, acc, false, false);
        }
      }
      // Scatter C/D tiles to LDS. Lane l owns (M = r + 8*lhalf, N = l16).
      const int rowbase = mt * 16 + 8 * lhalf;
      if (p < 2) {
        // Row-major [t][h]: rows stride by H_DIM -> b16 stores, branch-free.
        const int base = rowbase * H_DIM + l16;
#pragma unroll
        for (int nt = 0; nt < 4; ++nt) {
          const v8f acc = (nt == 0) ? acc0 : (nt == 1) ? acc1 : (nt == 2) ? acc2 : acc3;
#pragma unroll
          for (int r = 0; r < 8; ++r)
            qkDst[base + nt * 16 + r * H_DIM] = (__bf16)acc[r];
        }
      } else {
        // V^T [h][t]: the 8 rows are contiguous in t -> one b128 store per tile.
#pragma unroll
        for (int nt = 0; nt < 4; ++nt) {
          const v8f acc = (nt == 0) ? acc0 : (nt == 1) ? acc1 : (nt == 2) ? acc2 : acc3;
          v8bf pk;
#pragma unroll
          for (int r = 0; r < 8; ++r) pk[r] = (__bf16)acc[r];
          *(v8bf*)(&sVt[(nt * 16 + l16) * T_SEQ + rowbase]) = pk;
        }
      }
    }
    __syncthreads();
  }

  // ---------------- Phase 2: causal attention ----------------
  const float scale = 0.05103103630798288f;  // 1/sqrt(C=384)
#pragma unroll 1
  for (int r2 = 0; r2 < 2; ++r2) {
    const int mt = wave * 2 + r2;
    const int rowbase = 8 * lhalf;  // row-in-tile base for C/D tiles

    // Q A-fragments for both H-chunks (H=64 -> 2 WMMA k-steps)
    const __bf16* qrow = &sQ[(mt * 16 + l16) * H_DIM + kloA];
    v16bf aQ0 = pack16(*(const v8bf*)(qrow),      *(const v8bf*)(qrow + 16));
    v16bf aQ1 = pack16(*(const v8bf*)(qrow + 32), *(const v8bf*)(qrow + 48));

    // S = Q K^T (causal tiles only; guards are wave-uniform, non-causal = 0)
    v8f sT[16];
#pragma unroll
    for (int st = 0; st < 16; ++st) {
      if (st <= mt) {
        const __bf16* krow = &sK[(st * 16 + l16) * H_DIM + kloB];
        v16bf b0 = pack16(*(const v8bf*)(krow),      *(const v8bf*)(krow + 16));
        v16bf b1 = pack16(*(const v8bf*)(krow + 32), *(const v8bf*)(krow + 48));
        v8f c = v8f_zero();
        c = __builtin_amdgcn_wmma_f32_16x16x32_bf16(false, aQ0, false, b0, (short)0, c, false, false);
        c = __builtin_amdgcn_wmma_f32_16x16x32_bf16(false, aQ1, false, b1, (short)0, c, false, false);
        if (st == mt) {  // in-tile causal mask: col(l16) > row(r + 8*lhalf)
#pragma unroll
          for (int r = 0; r < 8; ++r)
            c[r] = (l16 > r + rowbase) ? -1e30f : c[r];
        }
        sT[st] = c;
      } else {
        sT[st] = v8f_zero();  // also serves as k-padding for P@V
      }
    }

    // Row-wise softmax (rows live per-VGPR; reduce across 16-lane halves)
    float rmax[8], rinv[8];
#pragma unroll
    for (int r = 0; r < 8; ++r) {
      float m = -1e30f;
#pragma unroll
      for (int st = 0; st < 16; ++st)
        if (st <= mt) m = fmaxf(m, sT[st][r]);
      m = fmaxf(m, __shfl_xor(m, 1));
      m = fmaxf(m, __shfl_xor(m, 2));
      m = fmaxf(m, __shfl_xor(m, 4));
      m = fmaxf(m, __shfl_xor(m, 8));
      rmax[r] = m;
    }
#pragma unroll
    for (int r = 0; r < 8; ++r) {
      float s = 0.0f;
#pragma unroll
      for (int st = 0; st < 16; ++st) {
        if (st <= mt) {
          float e = __expf((sT[st][r] - rmax[r]) * scale);
          sT[st][r] = e;
          s += e;
        }
      }
      s += __shfl_xor(s, 1);
      s += __shfl_xor(s, 2);
      s += __shfl_xor(s, 4);
      s += __shfl_xor(s, 8);
      rinv[r] = 1.0f / s;   // applied at O store (keeps P stores mul-free)
    }

    // Write unnormalized P (bf16) to per-wave LDS scratch (A-layout transpose).
    // Store all tiles covered by the 32-wide k-chunks; tile mt+1 is zeros.
    __bf16* sPw = sU.p[wave];
    const int nkc = (mt + 2) >> 1;
#pragma unroll
    for (int st = 0; st < 16; ++st) {
      if (st < 2 * nkc) {
#pragma unroll
        for (int r = 0; r < 8; ++r)
          sPw[(r + rowbase) * T_SEQ + st * 16 + l16] = (__bf16)sT[st][r];
      }
    }
    __syncthreads();  // uniform: all waves run both rounds

    // O = P V  (k trimmed to causal extent, rounded to 32)
    v8f o0 = v8f_zero(), o1 = v8f_zero(), o2 = v8f_zero(), o3 = v8f_zero();
    const __bf16* prow = &sPw[l16 * T_SEQ + kloA];
#pragma unroll
    for (int kc = 0; kc < 8; ++kc) {
      if (kc < nkc) {
        v16bf aP = pack16(*(const v8bf*)(prow + kc * 32),
                          *(const v8bf*)(prow + kc * 32 + 16));
#pragma unroll
        for (int nt = 0; nt < 4; ++nt) {
          const __bf16* vrow = &sVt[(nt * 16 + l16) * T_SEQ + kc * 32 + kloB];
          v16bf bb = pack16(*(const v8bf*)(vrow), *(const v8bf*)(vrow + 16));
          v8f& o = (nt == 0) ? o0 : (nt == 1) ? o1 : (nt == 2) ? o2 : o3;
          o = __builtin_amdgcn_wmma_f32_16x16x32_bf16(false, aP, false, bb,
                                                      (short)0, o, false, false);
        }
      }
    }

    // Store O tiles (fp32), applying the softmax normalization per row.
    float* orow = out + ((size_t)b * T_SEQ + mt * 16) * H_DIM;
#pragma unroll
    for (int nt = 0; nt < 4; ++nt) {
      const v8f o = (nt == 0) ? o0 : (nt == 1) ? o1 : (nt == 2) ? o2 : o3;
#pragma unroll
      for (int r = 0; r < 8; ++r)
        orow[(r + rowbase) * H_DIM + nt * 16 + l16] = o[r] * rinv[r];
    }
  }
}

extern "C" void kernel_launch(void* const* d_in, const int* in_sizes, int n_in,
                              void* d_out, int out_size, void* d_ws, size_t ws_size,
                              hipStream_t stream) {
  (void)in_sizes; (void)n_in; (void)d_ws; (void)ws_size; (void)out_size;
  const float* x  = (const float*)d_in[0];
  const float* Wq = (const float*)d_in[1];
  const float* Wk = (const float*)d_in[2];
  const float* Wv = (const float*)d_in[3];
  float* out = (float*)d_out;
  hipLaunchKernelGGL(attn_head_kernel, dim3(1024), dim3(256), 0, stream,
                     x, Wq, Wk, Wv, out);
}
